// DP_DIL_CNN_Block_V3_66228395704685
// MI455X (gfx1250) — matrix operand order repaired and checked by reading, not compiled, not run on hardware
//
#include <hip/hip_runtime.h>
#include <hip/hip_bf16.h>

// ---------------------------------------------------------------------------
// DP dual-path RNN block for MI455X (gfx1250, wave32, WMMA bf16).
// All dense math (conv1x1, dilated causal convs as 3 shifted GEMMs, LSTM input
// projections, FC layers, LSTM recurrent matmuls) runs on
// v_wmma_f32_16x16x32_bf16 with f32 accumulation. K is 128 everywhere, so the
// GEMM k-loop is fully unrolled and all fragment addresses fold to immediate
// offsets. LSTM scans are persistent one-wave-per-16-sequences kernels with
// compile-time shapes: two walking pointers per wave, immediate-offset loads.
// Workspace requirement: ~1.75 GB (documented arena below).
// ---------------------------------------------------------------------------

typedef unsigned short u16;
typedef __attribute__((ext_vector_type(16))) __bf16 v16bf;
typedef __attribute__((ext_vector_type(2)))  __bf16 v2bf;
typedef __attribute__((ext_vector_type(2)))  float  v2f;
typedef __attribute__((ext_vector_type(8)))  float  v8f;
typedef __attribute__((ext_vector_type(8)))  u16    v8u;
typedef __attribute__((ext_vector_type(16))) u16    v16u;

union FragU { v16u u; v16bf bf; };

static __device__ inline v16bf make_frag(v8u lo, v8u hi) {
  FragU f;
  f.u = __builtin_shufflevector(lo, hi, 0,1,2,3,4,5,6,7,8,9,10,11,12,13,14,15);
  return f.bf;
}

// f32 -> bf16 via native fptrunc (RNE); packed form uses v_cvt_pk_bf16_f32.
static __device__ inline u16 f2bf(float f) {
  return __builtin_bit_cast(u16, (__bf16)f);
}
static __device__ inline unsigned pk2bf(float a, float b) {
#if __has_builtin(__builtin_amdgcn_cvt_pk_bf16_f32)
  auto r = __builtin_amdgcn_cvt_pk_bf16_f32(a, b);
  return __builtin_bit_cast(unsigned, r);
#else
  v2f v; v.x = a; v.y = b;
  return __builtin_bit_cast(unsigned, __builtin_convertvector(v, v2bf));
#endif
}

static __device__ inline float sigm(float x) { return 1.0f / (1.0f + __expf(-x)); }

// ---------------------------------------------------------------------------
// Weight packer: B[k][n] = W[k*sK + n*sN] -> packed bf16 fragment order.
// Tile (kt,nt) holds 512 values: lane L covers column n = nt*16 + (L&15),
// K range kt*32 + (L<16 ? 0..15 : 16..31), contiguous per lane (32B).
// ---------------------------------------------------------------------------
__global__ void pack_b(const float* __restrict__ W, u16* __restrict__ out,
                       int K, int N, int sK, int sN)
{
  int id = blockIdx.x * 256 + threadIdx.x;
  if (id >= K * N) return;
  int tile = id >> 9;
  int rem  = id & 511;
  int L    = rem >> 4;
  int v    = rem & 15;
  int kTiles = K >> 5;
  int nt = tile / kTiles;
  int kt = tile % kTiles;
  int n = nt * 16 + (L & 15);
  int k = kt * 32 + ((L >> 4) << 4) + v;
  out[id] = f2bf(W[(long)k * sK + (long)n * sN]);
}

// ---------------------------------------------------------------------------
// Tiled GEMM (K fixed at 128): C[M,N] = act( beta*C + A[M,128] @ Bp + bias ).
// Block = 256 threads (8 waves) -> 64x128 tile; wave = 16x64 (4 accumulators).
// "shift/seg": logical A row r reads physical row r-shift if (r%seg)>=shift
// else zero-row (causal dilated conv taps). act: 0 none, 1 sigmoid, 2 prelu.
// Requires M%64==0, N%128==0 (true for every call site here).
// ---------------------------------------------------------------------------
__global__ __launch_bounds__(256)
void gemm_wmma(const float* __restrict__ A, const u16* __restrict__ Bp,
               const float* __restrict__ bias, float* __restrict__ C,
               int M, int N, int beta, int act,
               const float* __restrict__ alpha_p, int shift, int seg)
{
  constexpr int K = 128, kTiles = 4;
  __shared__ u16 As[64 * 40];                 // 64 rows x 32 k, pad->40 u16
  const int tid  = threadIdx.x;
  const int lane = tid & 31;
  const int wave = tid >> 5;
  const int rt   = wave & 3;                  // row tile within block
  const int cg   = wave >> 2;                 // column half (0/1)
  const int bx   = blockIdx.x;
  const int by   = blockIdx.y;
  const int row0 = by * 64;
  const int lm = lane & 15;
  const int lh = lane >> 4;

  v8f acc[4];
  #pragma unroll
  for (int i = 0; i < 4; ++i) {
    if (beta) {
      const int gn = bx * 128 + cg * 64 + i * 16 + lm;
      #pragma unroll
      for (int r = 0; r < 8; ++r) {
        int m = row0 + rt * 16 + lh * 8 + r;
        acc[i][r] = C[(long)m * N + gn];
      }
    } else {
      #pragma unroll
      for (int r = 0; r < 8; ++r) acc[i][r] = 0.0f;
    }
  }

  // per-thread staging coordinates (loop-invariant)
  const int sr  = tid >> 2;                   // 0..63 : row within block
  const int skc = (tid & 3) * 8;              // 0,8,16,24 : k within slab
  const int sgr = row0 + sr;
  const bool svalid = (sgr % seg) >= shift;
  const float* srow = A + (long)(sgr - shift) * K + skc;
  unsigned* sdst = (unsigned*)&As[sr * 40 + skc];
  // loop-invariant base for this wave's B fragments; tile offsets are
  // compile-time immediates after full unroll
  const u16* bpBase = Bp + (long)(bx * 8 + cg * 4) * (kTiles * 512) + lane * 16;
  // A-fragment LDS base
  const u16* afBase = &As[(rt * 16 + lm) * 40 + lh * 16];

  #pragma unroll
  for (int kt = 0; kt < kTiles; ++kt) {
    __syncthreads();
    if (svalid) {                             // stage A slab (f32 -> bf16)
      const float* src = srow + kt * 32;
      float4 a0 = *(const float4*)(src);
      float4 a1 = *(const float4*)(src + 4);
      sdst[0] = pk2bf(a0.x, a0.y);
      sdst[1] = pk2bf(a0.z, a0.w);
      sdst[2] = pk2bf(a1.x, a1.y);
      sdst[3] = pk2bf(a1.z, a1.w);
      if (kt + 1 < kTiles)                    // global_prefetch of next slab
        __builtin_prefetch(srow + (kt + 1) * 32, 0, 3);
    } else {
      sdst[0] = 0u; sdst[1] = 0u; sdst[2] = 0u; sdst[3] = 0u;
    }
    __syncthreads();

    v8u alo = *(const v8u*)(afBase);
    v8u ahi = *(const v8u*)(afBase + 8);
    v16bf afrag = make_frag(alo, ahi);

    #pragma unroll
    for (int i = 0; i < 4; ++i) {
      const u16* bp = bpBase + (i * kTiles + kt) * 512;   // immediate offset
      v8u blo = *(const v8u*)(bp);
      v8u bhi = *(const v8u*)(bp + 8);
      acc[i] = __builtin_amdgcn_wmma_f32_16x16x32_bf16(
          false, afrag, false, make_frag(blo, bhi), (short)0, acc[i],
          false, false);
    }
  }

  float alpha = alpha_p ? *alpha_p : 0.0f;
  #pragma unroll
  for (int i = 0; i < 4; ++i) {
    int gn = bx * 128 + cg * 64 + i * 16 + lm;
    float bv = bias ? bias[gn] : 0.0f;
    #pragma unroll
    for (int r = 0; r < 8; ++r) {
      float v = acc[i][r] + bv;
      if (act == 1)      v = sigm(v);
      else if (act == 2) v = (v >= 0.0f) ? v : alpha * v;
      int m = row0 + rt * 16 + lh * 8 + r;
      C[(long)m * N + gn] = v;
    }
  }
}

// ---------------------------------------------------------------------------
// Persistent LSTM scan (fully compile-time shapes). One wave / 16 sequences.
// Gate order i,f,g,o. xg: input projection incl. biases, [NS, L, 4*HH].
// Per step: g = xg[t] + h @ Whh^T via WMMA; c,h in-register; h staged through
// a wave-private LDS tile to rebuild A-fragments. Whh streams from L2.
// Two walking pointers (xg, hOut) advance +/- one step per iteration; all
// row/gate offsets are immediates. Single-wave workgroup => barriers are NOPs.
// ---------------------------------------------------------------------------
template<int HHt, bool REV, int L, int OS>
__global__ __launch_bounds__(32)
void lstm_scan(const float* __restrict__ xg, const u16* __restrict__ WhhP,
               float* __restrict__ hOut, int colOff)
{
  constexpr int G4 = 4 * HHt;
  constexpr int KT = HHt / 32;
  constexpr int JT = HHt / 16;
  constexpr int t0 = REV ? (L - 1) : 0;
  constexpr int xstep = REV ? -G4 : G4;
  constexpr int hstep = REV ? -OS : OS;
  __shared__ u16 hlds[16 * HHt];
  const int lane = threadIdx.x;
  const int lm = lane & 15, lh = lane >> 4;
  const int n0 = blockIdx.x * 16;

  for (int i = lane; i < 16 * HHt; i += 32) hlds[i] = 0;
  v8f c[JT];
  #pragma unroll
  for (int j = 0; j < JT; ++j)
    #pragma unroll
    for (int r = 0; r < 8; ++r) c[j][r] = 0.0f;
  __syncthreads();

  const float* xgq = xg + ((long)(n0 + lh * 8) * L + t0) * G4 + lm;
  float*       hq  = hOut + ((long)(n0 + lh * 8) * L + t0) * OS + colOff + lm;
  const u16*   wq  = WhhP + lane * 16;
  const u16*   hfB = &hlds[lm * HHt + lh * 16];

  for (int s = 0; s < L; ++s) {
    v16bf afr[KT];
    #pragma unroll
    for (int kt = 0; kt < KT; ++kt) {
      v8u lo = *(const v8u*)(hfB + kt * 32);
      v8u hi = *(const v8u*)(hfB + kt * 32 + 8);
      afr[kt] = make_frag(lo, hi);
    }
    __syncthreads();   // frags read before h rewrite below

    #pragma unroll
    for (int j = 0; j < JT; ++j) {
      v8f g[4];
      #pragma unroll
      for (int gi = 0; gi < 4; ++gi) {
        #pragma unroll
        for (int r = 0; r < 8; ++r)
          g[gi][r] = xgq[(long)r * (L * G4) + gi * HHt + j * 16];
      }
      #pragma unroll
      for (int kt = 0; kt < KT; ++kt) {
        #pragma unroll
        for (int gi = 0; gi < 4; ++gi) {
          const u16* bp = wq + ((gi * JT + j) * KT + kt) * 512;  // immediate
          v8u blo = *(const v8u*)bp;
          v8u bhi = *(const v8u*)(bp + 8);
          g[gi] = __builtin_amdgcn_wmma_f32_16x16x32_bf16(
              false, afr[kt], false, make_frag(blo, bhi), (short)0, g[gi],
              false, false);
        }
      }
      #pragma unroll
      for (int r = 0; r < 8; ++r) {
        float iv = sigm(g[0][r]);
        float fv = sigm(g[1][r]);
        float gv = tanhf(g[2][r]);
        float ov = sigm(g[3][r]);
        float cv = fv * c[j][r] + iv * gv;
        c[j][r] = cv;
        float hv = ov * tanhf(cv);
        hlds[(lh * 8 + r) * HHt + j * 16 + lm] = f2bf(hv);
        hq[(long)r * (L * OS) + j * 16] = hv;
      }
    }
    __syncthreads();   // new h visible before next step's frag loads
    xgq += xstep;
    hq  += hstep;
  }
}

// ---------------------------------------------------------------------------
// LayerNorm over last two dims (F,C). in: (B,T,F,C); out: (B,C,T,F),
// optional residual (B,C,T,F). One block per (b,t).
// ---------------------------------------------------------------------------
__global__ __launch_bounds__(256)
void layernorm2_k(const float* __restrict__ in, const float* __restrict__ w,
                  const float* __restrict__ b, const float* __restrict__ res,
                  float* __restrict__ out, int T_, int F_, int C_)
{
  __shared__ float red[256];
  int bt = blockIdx.x;
  int bb = bt / T_, t = bt % T_;
  const float* x = in + (long)bt * F_ * C_;
  int NE = F_ * C_;
  float s = 0.0f;
  for (int i = threadIdx.x; i < NE; i += 256) s += x[i];
  red[threadIdx.x] = s; __syncthreads();
  for (int o = 128; o > 0; o >>= 1) {
    if (threadIdx.x < o) red[threadIdx.x] += red[threadIdx.x + o];
    __syncthreads();
  }
  float mean = red[0] / NE; __syncthreads();
  float v = 0.0f;
  for (int i = threadIdx.x; i < NE; i += 256) { float d = x[i] - mean; v += d * d; }
  red[threadIdx.x] = v; __syncthreads();
  for (int o = 128; o > 0; o >>= 1) {
    if (threadIdx.x < o) red[threadIdx.x] += red[threadIdx.x + o];
    __syncthreads();
  }
  float rstd = rsqrtf(red[0] / NE + 1e-5f);
  for (int i = threadIdx.x; i < NE; i += 256) {
    int f = i / C_, cch = i % C_;
    float y = (x[i] - mean) * rstd * w[i] + b[i];
    long oidx = (((long)bb * C_ + cch) * T_ + t) * F_ + f;
    if (res) y += res[oidx];
    out[oidx] = y;
  }
}

// ------------------------- layout / elementwise ----------------------------
__global__ void perm_bctf_to_btfc(const float* __restrict__ in, float* __restrict__ out,
                                  int Cc, int Tt, int Ff, long n)
{
  long i = (long)blockIdx.x * 256 + threadIdx.x;
  if (i >= n) return;
  int cch = (int)(i % Cc); long r = i / Cc;
  int f = (int)(r % Ff); r /= Ff;
  int t = (int)(r % Tt); long bb = r / Tt;
  out[i] = in[(((long)bb * Cc + cch) * Tt + t) * Ff + f];
}
__global__ void perm_bctf_to_bftc(const float* __restrict__ in, float* __restrict__ out,
                                  int Cc, int Tt, int Ff, long n)
{
  long i = (long)blockIdx.x * 256 + threadIdx.x;
  if (i >= n) return;
  int cch = (int)(i % Cc); long r = i / Cc;
  int t = (int)(r % Tt); r /= Tt;
  int f = (int)(r % Ff); long bb = r / Ff;
  out[i] = in[(((long)bb * Cc + cch) * Tt + t) * Ff + f];
}
// rows (b,d1,d2,c) -> rows (b,d2,d1,c)
__global__ void perm_swap12(const float* __restrict__ in, float* __restrict__ out,
                            int D1, int D2, int Cc, long n)
{
  long i = (long)blockIdx.x * 256 + threadIdx.x;
  if (i >= n) return;
  int cch = (int)(i % Cc); long r = i / Cc;
  int d1 = (int)(r % D1); r /= D1;
  int d2 = (int)(r % D2); long bb = r / D2;
  out[i] = in[(((long)bb * D1 + d1) * D2 + d2) * Cc + cch];
}
__global__ void prelu_k(const float* __restrict__ in, float* __restrict__ out,
                        const float* __restrict__ ap, long n)
{
  long i = (long)blockIdx.x * 256 + threadIdx.x;
  if (i >= n) return;
  float a = *ap;
  float x = in[i];
  out[i] = (x >= 0.0f) ? x : a * x;
}
__global__ void mul_k(const float* __restrict__ a, const float* __restrict__ b,
                      float* __restrict__ out, long n)
{
  long i = (long)blockIdx.x * 256 + threadIdx.x;
  if (i >= n) return;
  out[i] = a[i] * b[i];
}
__global__ void acc_k(float* __restrict__ dst, const float* __restrict__ src,
                      long n, int first)
{
  long i = (long)blockIdx.x * 256 + threadIdx.x;
  if (i >= n) return;
  dst[i] = first ? src[i] : (dst[i] + src[i]);
}
__global__ void add3_k(const float* __restrict__ a, const float* __restrict__ b,
                       const float* __restrict__ cc, float* __restrict__ out, long n)
{
  long i = (long)blockIdx.x * 256 + threadIdx.x;
  if (i >= n) return;
  out[i] = a[i] + b[i] + cc[i];
}
__global__ void addv_k(const float* __restrict__ a, const float* __restrict__ b,
                       float* __restrict__ out, int n)
{
  int i = blockIdx.x * 256 + threadIdx.x;
  if (i < n) out[i] = a[i] + b[i];
}

// ---------------------------------------------------------------------------
extern "C" void kernel_launch(void* const* d_in, const int* in_sizes, int n_in,
                              void* d_out, int out_size, void* d_ws, size_t ws_size,
                              hipStream_t stream)
{
  (void)in_sizes; (void)n_in; (void)out_size; (void)ws_size;
  const int Bn = 8, Cn = 128, Tn = 250, Fn = 128, Hn = 64;
  const long MTF = (long)Bn * Tn * Fn;      // 256000 token rows
  const long SZ1 = MTF * Cn;                // 32,768,000 floats
  const long SZ2 = MTF * 256;
  const long SZ4 = MTF * 512;

  const float* x = (const float*)d_in[0];
  const float* iwf_ih = (const float*)d_in[1];
  const float* iwf_hh = (const float*)d_in[2];
  const float* ibf_ih = (const float*)d_in[3];
  const float* ibf_hh = (const float*)d_in[4];
  const float* iwr_ih = (const float*)d_in[5];
  const float* iwr_hh = (const float*)d_in[6];
  const float* ibr_ih = (const float*)d_in[7];
  const float* ibr_hh = (const float*)d_in[8];
  const float* intra_fc_w = (const float*)d_in[9];
  const float* intra_fc_b = (const float*)d_in[10];
  const float* intra_ln_w = (const float*)d_in[11];
  const float* intra_ln_b = (const float*)d_in[12];
  const float* ew_ih = (const float*)d_in[13];
  const float* ew_hh = (const float*)d_in[14];
  const float* eb_ih = (const float*)d_in[15];
  const float* eb_hh = (const float*)d_in[16];
  const float* inter_fc_w = (const float*)d_in[17];
  const float* inter_fc_b = (const float*)d_in[18];
  const float* inter_ln_w = (const float*)d_in[19];
  const float* inter_ln_b = (const float*)d_in[20];
  const float* freq_ln_w = (const float*)d_in[21];
  const float* freq_ln_b = (const float*)d_in[22];

  // ---- workspace arena (region reuse across pipeline phases) ----
  float* ws    = (float*)d_ws;
  float* gates = ws;                 // SZ4: xg_fwd | xg_rev, later inter gates
  float* RA = ws + SZ4;              // xi -> h2 -> lbuf
  float* RB = RA + SZ1;              // hcat -> d2p -> rbuf
  float* RC = RB + SZ1;              // d1 -> d2q -> yin
  float* RD = RC + SZ1;              // intra_out (live to end)
  float* RE = RD + SZ1;              // xj/fi -> foT
  float* RF = RE + SZ1;              // inter_out (live to end)
  float* RG = RF + SZ1;              // prelu temp
  float* RH = RG + SZ1;              // glu_out -> fn
  float* RI = RH + SZ1;              // skip accumulator
  float* misc = RI + SZ1;            // biases + packed bf16 weights

  float* bias_if = misc;             // 256
  float* bias_ir = misc + 256;       // 256
  float* bias_e  = misc + 512;       // 512
  u16* pk = (u16*)(misc + 1024);
  u16* pk_iwf = pk; pk += 128 * 256;
  u16* pk_iwr = pk; pk += 128 * 256;
  u16* pk_whf = pk; pk += 64 * 256;
  u16* pk_whr = pk; pk += 64 * 256;
  u16* pk_fc1 = pk; pk += 128 * 128;
  u16* pk_ew  = pk; pk += 128 * 512;
  u16* pk_ehh = pk; pk += 128 * 512;
  u16* pk_fc2 = pk; pk += 128 * 128;
  u16* pk_glu[5][8];
  for (int i = 0; i < 5; ++i)
    for (int j = 0; j < 8; ++j) { pk_glu[i][j] = pk; pk += 128 * 128; }

  auto pack = [&](const float* W, u16* out, int K, int N, int sK, int sN) {
    int tot = K * N;
    pack_b<<<(tot + 255) / 256, 256, 0, stream>>>(W, out, K, N, sK, sN);
  };
  auto gemm = [&](const float* A, const u16* Bp, const float* bias, float* Cm,
                  int M, int N, int beta, int act, const float* alpha,
                  int shift, int seg) {
    dim3 g(N / 128, M / 64);
    gemm_wmma<<<g, 256, 0, stream>>>(A, Bp, bias, Cm, M, N, beta, act,
                                     alpha, shift, seg);
  };
  const long NE = SZ1;
  const int EB = (int)((NE + 255) / 256);

  // ---- pack all weights (W stored [out,in] row-major -> B[k][n]=W[n][k]) ----
  pack(iwf_ih, pk_iwf, 128, 256, 1, 128);
  pack(iwr_ih, pk_iwr, 128, 256, 1, 128);
  pack(iwf_hh, pk_whf, 64, 256, 1, 64);
  pack(iwr_hh, pk_whr, 64, 256, 1, 64);
  pack(intra_fc_w, pk_fc1, 128, 128, 1, 128);
  pack(ew_ih, pk_ew, 128, 512, 1, 128);
  pack(ew_hh, pk_ehh, 128, 512, 1, 128);
  pack(inter_fc_w, pk_fc2, 128, 128, 1, 128);
  for (int i = 0; i < 5; ++i) {
    const int gb = 23 + 11 * i;
    pack((const float*)d_in[gb + 0], pk_glu[i][0], 128, 128, 1, 128);     // in_w
    for (int k = 0; k < 3; ++k)                                           // wl taps
      pack((const float*)d_in[gb + 3] + k, pk_glu[i][1 + k], 128, 128, 3, 384);
    for (int k = 0; k < 3; ++k)                                           // wr taps
      pack((const float*)d_in[gb + 6] + k, pk_glu[i][4 + k], 128, 128, 3, 384);
    pack((const float*)d_in[gb + 8], pk_glu[i][7], 128, 128, 1, 128);     // out_w
  }
  addv_k<<<1, 256, 0, stream>>>(ibf_ih, ibf_hh, bias_if, 256);
  addv_k<<<1, 256, 0, stream>>>(ibr_ih, ibr_hh, bias_ir, 256);
  addv_k<<<2, 256, 0, stream>>>(eb_ih, eb_hh, bias_e, 512);

  // ================= intra path (BiLSTM over frequency) =================
  perm_bctf_to_btfc<<<EB, 256, 0, stream>>>(x, RA, Cn, Tn, Fn, NE);     // xi
  gemm(RA, pk_iwf, bias_if, gates,       (int)MTF, 256, 0, 0, nullptr, 0, 1);
  gemm(RA, pk_iwr, bias_ir, gates + SZ2, (int)MTF, 256, 0, 0, nullptr, 0, 1);
  lstm_scan<64, false, 128, 128><<<125, 32, 0, stream>>>(gates,       pk_whf, RB, 0);
  lstm_scan<64, true,  128, 128><<<125, 32, 0, stream>>>(gates + SZ2, pk_whr, RB, Hn);
  gemm(RB, pk_fc1, intra_fc_b, RC, (int)MTF, 128, 0, 0, nullptr, 0, 1); // d1
  layernorm2_k<<<Bn * Tn, 256, 0, stream>>>(RC, intra_ln_w, intra_ln_b, x, RD,
                                            Tn, Fn, Cn);                // intra_out

  // ================= inter path (LSTM over time) =================
  perm_bctf_to_bftc<<<EB, 256, 0, stream>>>(RD, RE, Cn, Tn, Fn, NE);    // xj / fi
  gemm(RE, pk_ew, bias_e, gates, (int)MTF, 512, 0, 0, nullptr, 0, 1);
  lstm_scan<128, false, 250, 128><<<64, 32, 0, stream>>>(gates, pk_ehh, RA, 0); // h2
  gemm(RA, pk_fc2, inter_fc_b, RB, (int)MTF, 128, 0, 0, nullptr, 0, 1);  // d2p
  perm_swap12<<<EB, 256, 0, stream>>>(RB, RC, Fn, Tn, Cn, NE);          // d2q (b,t,f,c)
  layernorm2_k<<<Bn * Tn, 256, 0, stream>>>(RC, inter_ln_w, inter_ln_b, nullptr,
                                            RF, Tn, Fn, Cn);            // inter_out

  // ================= freq path (5 dilated causal GLU blocks) =================
  for (int i = 0; i < 5; ++i) {
    const int gb = 23 + 11 * i;
    const float* in_b  = (const float*)d_in[gb + 1];
    const float* a_l   = (const float*)d_in[gb + 2];
    const float* bl    = (const float*)d_in[gb + 4];
    const float* a_r   = (const float*)d_in[gb + 5];
    const float* br    = (const float*)d_in[gb + 7];
    const float* out_b = (const float*)d_in[gb + 9];
    const float* a_o   = (const float*)d_in[gb + 10];
    const int dila = 1 << i;

    gemm(RE, pk_glu[i][0], in_b, RC, (int)MTF, 128, 0, 0, nullptr, 0, 1); // yin
    prelu_k<<<EB, 256, 0, stream>>>(RC, RG, a_l, NE);
    for (int k = 0; k < 3; ++k)                                // causal dilated conv L
      gemm(RG, pk_glu[i][1 + k], (k == 0) ? bl : nullptr, RA, (int)MTF, 128,
           (k == 0) ? 0 : 1, 0, nullptr, (2 - k) * dila, Tn);
    prelu_k<<<EB, 256, 0, stream>>>(RC, RG, a_r, NE);
    for (int k = 0; k < 3; ++k)                                // conv R (+sigmoid)
      gemm(RG, pk_glu[i][4 + k], (k == 0) ? br : nullptr, RB, (int)MTF, 128,
           (k == 0) ? 0 : 1, (k == 2) ? 1 : 0, nullptr, (2 - k) * dila, Tn);
    mul_k<<<EB, 256, 0, stream>>>(RA, RB, RA, NE);             // l * sigmoid(r)
    gemm(RA, pk_glu[i][7], out_b, RH, (int)MTF, 128, 0, 2, a_o, 0, 1);
    acc_k<<<EB, 256, 0, stream>>>(RI, RH, NE, (i == 0) ? 1 : 0);   // skip sum
  }
  perm_swap12<<<EB, 256, 0, stream>>>(RI, RE, Fn, Tn, Cn, NE);   // (b,t,f,c)
  layernorm2_k<<<Bn * Tn, 256, 0, stream>>>(RE, freq_ln_w, freq_ln_b, nullptr,
                                            RH, Tn, Fn, Cn);     // fn (B,C,T,F)

  // ================= final sum =================
  add3_k<<<EB, 256, 0, stream>>>(RF, RD, RH, (float*)d_out, NE);
}
